// QuantumTransformerBlock_65481071408117
// MI455X (gfx1250) — compile-verified, matrix operand order checked by reading, not run on hardware
//
#include <hip/hip_runtime.h>
#include <hip/hip_bf16.h>
#include <math.h>

typedef __attribute__((ext_vector_type(16))) _Float16 v16h;
typedef __attribute__((ext_vector_type(8)))  _Float16 v8h;
typedef __attribute__((ext_vector_type(4)))  _Float16 v4h;
typedef __attribute__((ext_vector_type(8)))  float    v8f;

#define WMMA_F16(a, b, c) \
  __builtin_amdgcn_wmma_f32_16x16x32_f16(false, (a), false, (b), (short)0, (c), false, false)

// Problem constants (B=2, S=2048, E=1024, H=16, DK=64, FFN=4096, NQ=4)
static constexpr int kM  = 4096;   // B*S
static constexpr int kE  = 1024;
static constexpr int kH  = 16;
static constexpr int kDK = 64;
static constexpr int kS  = 2048;

// ---------------------------------------------------------------------------
// 16x32 f16 A-fragment loader (also used for B^T fragments).
// ISA layout: lanes 0-15 = rows 0-15; K-halves split by lane[4]:
//   lanes 0-15:  VGPR0-3 = K 0..7,  VGPR4-7 = K 16..23
//   lanes 16-31: VGPR0-3 = K 8..15, VGPR4-7 = K 24..31
// -> per lane: two contiguous 16B runs at k = 8*(lane>>4) and k+16.
// ---------------------------------------------------------------------------
__device__ inline v16h frag_ld(const _Float16* __restrict__ tile, int ld, int lane) {
  const _Float16* p = tile + (lane & 15) * ld + ((lane >> 4) << 3);
  v8h lo = *(const v8h*)(p);
  v8h hi = *(const v8h*)(p + 16);
  return __builtin_shufflevector(lo, hi, 0,1,2,3,4,5,6,7,8,9,10,11,12,13,14,15);
}

// ---------------------------------------------------------------------------
// f32 -> f16 convert (4 elems / thread)
// ---------------------------------------------------------------------------
__global__ __launch_bounds__(256) void cvt_f32_f16(const float* __restrict__ in,
                                                   _Float16* __restrict__ out, int n) {
  int i = (blockIdx.x * 256 + threadIdx.x) * 4;
  if (i < n) {
    float4 v = *(const float4*)(in + i);
    v4h h = {(_Float16)v.x, (_Float16)v.y, (_Float16)v.z, (_Float16)v.w};
    *(v4h*)(out + i) = h;
  }
}

// ---------------------------------------------------------------------------
// Fused convert + transpose: in f32 [R x C] -> out f16 [C x R].
// LDS-tiled (64x64) so both global sides are fully vectorized.
// ---------------------------------------------------------------------------
__global__ __launch_bounds__(256) void cvt_transpose_f32f16(const float* __restrict__ in,
                                                            _Float16* __restrict__ out,
                                                            int R, int C) {
  __shared__ __align__(16) _Float16 tile[64][72];  // [c][r]
  const int t = threadIdx.x;
  const int r0 = blockIdx.y * 64, c0 = blockIdx.x * 64;
#pragma unroll
  for (int i = 0; i < 4; ++i) {       // 64x64 f32 = 1024 float4 chunks
    int c = t + 256 * i;
    int rr = c >> 4, cc = (c & 15) << 2;
    float4 v = *(const float4*)&in[(size_t)(r0 + rr) * C + c0 + cc];
    tile[cc + 0][rr] = (_Float16)v.x;
    tile[cc + 1][rr] = (_Float16)v.y;
    tile[cc + 2][rr] = (_Float16)v.z;
    tile[cc + 3][rr] = (_Float16)v.w;
  }
  __syncthreads();
#pragma unroll
  for (int i = 0; i < 2; ++i) {       // 64x64 f16 = 512 v8h chunks
    int c = t + 256 * i;
    int cr = c >> 3, rk = (c & 7) << 3;
    *(v8h*)&out[(size_t)(c0 + cr) * R + r0 + rk] = *(const v8h*)&tile[cr][rk];
  }
}

// ---------------------------------------------------------------------------
// f16 transpose: in [R x C] -> out [C x R] (for V -> V^T), LDS-tiled 64x64.
// ---------------------------------------------------------------------------
__global__ __launch_bounds__(256) void transpose_f16(const _Float16* __restrict__ in,
                                                     _Float16* __restrict__ out,
                                                     int R, int C) {
  __shared__ __align__(16) _Float16 tile[64][72];  // [c][r]
  const int t = threadIdx.x;
  const int r0 = blockIdx.y * 64, c0 = blockIdx.x * 64;
#pragma unroll
  for (int i = 0; i < 2; ++i) {
    int c = t + 256 * i;
    int rr = c >> 3, cc = (c & 7) << 3;
    v8h v = *(const v8h*)&in[(size_t)(r0 + rr) * C + c0 + cc];
#pragma unroll
    for (int q = 0; q < 8; ++q) tile[cc + q][rr] = v[q];
  }
  __syncthreads();
#pragma unroll
  for (int i = 0; i < 2; ++i) {
    int c = t + 256 * i;
    int cr = c >> 3, rk = (c & 7) << 3;
    *(v8h*)&out[(size_t)(c0 + cr) * R + r0 + rk] = *(const v8h*)&tile[cr][rk];
  }
}

// ---------------------------------------------------------------------------
// Tiled f16 GEMM, f32 accum: C[M,N] = A[M,K] * W[K,N], W given TRANSPOSED
// (Wt = [N x K]) so all LDS fills and fragment loads are b128.
// Block: 256 thr (8 waves, 2x4). Block tile 128x256, BK=32, wave tile 64x64.
// Double-buffered LDS: stage k+1 in regs during WMMAs on k; 1 barrier/iter.
// ---------------------------------------------------------------------------
template <bool OUT_F16>
__global__ __launch_bounds__(256) void gemm_wmma(const _Float16* __restrict__ A,
                                                 const _Float16* __restrict__ Wt,
                                                 _Float16* __restrict__ Oh,
                                                 float* __restrict__ Of,
                                                 int M, int N, int K) {
  __shared__ __align__(16) _Float16 lA[2][128][40];  // [m][k]
  __shared__ __align__(16) _Float16 lB[2][256][40];  // [n][k]
  const int t = threadIdx.x, lane = t & 31, wave = t >> 5;
  const int wm = wave >> 2, wn = wave & 3;           // 2x4 wave grid
  const int m0 = blockIdx.y * 128, n0 = blockIdx.x * 256;

  const v8f zero = {0.f, 0.f, 0.f, 0.f, 0.f, 0.f, 0.f, 0.f};
  v8f acc[4][4];
#pragma unroll
  for (int i = 0; i < 4; ++i)
#pragma unroll
    for (int j = 0; j < 4; ++j) acc[i][j] = zero;

  // prologue: tile 0 -> buffer 0
#pragma unroll
  for (int i = 0; i < 2; ++i) {
    int c = t + 256 * i;
    int r = c >> 2, col = (c & 3) << 3;
    *(v8h*)&lA[0][r][col] = *(const v8h*)&A[(size_t)(m0 + r) * K + col];
  }
#pragma unroll
  for (int i = 0; i < 4; ++i) {
    int c = t + 256 * i;
    int r = c >> 2, col = (c & 3) << 3;
    *(v8h*)&lB[0][r][col] = *(const v8h*)&Wt[(size_t)(n0 + r) * K + col];
  }
  __syncthreads();

  const int nk = K / 32;
  for (int kt = 0; kt < nk; ++kt) {
    const int cur = kt & 1, nxt = cur ^ 1;
    const bool has_next = (kt + 1 < nk);
    v8h sa[2], sb[4];
    if (has_next) {
      const int k0 = (kt + 1) * 32;
#pragma unroll
      for (int i = 0; i < 2; ++i) {
        int c = t + 256 * i;
        int r = c >> 2, col = (c & 3) << 3;
        sa[i] = *(const v8h*)&A[(size_t)(m0 + r) * K + k0 + col];
      }
#pragma unroll
      for (int i = 0; i < 4; ++i) {
        int c = t + 256 * i;
        int r = c >> 2, col = (c & 3) << 3;
        sb[i] = *(const v8h*)&Wt[(size_t)(n0 + r) * K + k0 + col];
      }
    }

    v16h af[4];
#pragma unroll
    for (int i = 0; i < 4; ++i) af[i] = frag_ld(&lA[cur][wm * 64 + 16 * i][0], 40, lane);
#pragma unroll
    for (int j = 0; j < 4; ++j) {
      v16h bf = frag_ld(&lB[cur][wn * 64 + 16 * j][0], 40, lane);
#pragma unroll
      for (int i = 0; i < 4; ++i) acc[i][j] = WMMA_F16(af[i], bf, acc[i][j]);
    }

    if (has_next) {
#pragma unroll
      for (int i = 0; i < 2; ++i) {
        int c = t + 256 * i;
        int r = c >> 2, col = (c & 3) << 3;
        *(v8h*)&lA[nxt][r][col] = sa[i];
      }
#pragma unroll
      for (int i = 0; i < 4; ++i) {
        int c = t + 256 * i;
        int r = c >> 2, col = (c & 3) << 3;
        *(v8h*)&lB[nxt][r][col] = sb[i];
      }
    }
    __syncthreads();
  }

  // C/D layout: VGPR r -> row r + 8*(lane>>4), col = lane&15 (per 16x16 frag)
  const int half = lane >> 4, ln = lane & 15;
#pragma unroll
  for (int i = 0; i < 4; ++i)
#pragma unroll
    for (int j = 0; j < 4; ++j)
#pragma unroll
      for (int r = 0; r < 8; ++r) {
        int row = m0 + wm * 64 + 16 * i + r + 8 * half;
        int col = n0 + wn * 64 + 16 * j + ln;
        float v = acc[i][j][r];
        if (OUT_F16) Oh[(size_t)row * N + col] = (_Float16)v;
        else         Of[(size_t)row * N + col] = v;
      }
}

// ---------------------------------------------------------------------------
// Flash attention (per b,h): block = 128 thr (4 waves), 64 query rows/block.
// V is pre-transposed globally (VhT = [E x M]) so K and V tile fills are b128.
// Adds c_attn = cos(qp[0]+qp[1]) to channels d<4 of each head ("mixed").
// ---------------------------------------------------------------------------
__global__ __launch_bounds__(128) void attn_wmma(const _Float16* __restrict__ Q,
                                                 const _Float16* __restrict__ Km,
                                                 const _Float16* __restrict__ VhT,
                                                 const float* __restrict__ qp_attn,
                                                 _Float16* __restrict__ Out) {
  __shared__ __align__(16) _Float16 Kt[64][72];     // [kseq][d]  (K^T frags)
  __shared__ __align__(16) _Float16 VtT[64][72];    // [d][kseq]  (V B-frags)
  __shared__ __align__(16) _Float16 Pl[4][16][72];  // per-wave P [qrow][kseq]
  const int t = threadIdx.x, lane = t & 31, wave = t >> 5;
  const int half = lane >> 4, ln = lane & 15;
  const int qt = blockIdx.x, bh = blockIdx.y;
  const int b = bh >> 4, h = bh & 15;

  const size_t rowQ = (size_t)(b * kS + qt * 64 + wave * 16);
  const _Float16* qbase = Q + rowQ * kE + h * kDK;
  v16h qa0 = frag_ld(qbase, kE, lane);        // K-dim 0..31
  v16h qa1 = frag_ld(qbase + 32, kE, lane);   // K-dim 32..63

  const v8f zero = {0.f, 0.f, 0.f, 0.f, 0.f, 0.f, 0.f, 0.f};
  v8f o[4];
  float m_run[8], l_run[8];
#pragma unroll
  for (int n = 0; n < 4; ++n) o[n] = zero;
#pragma unroll
  for (int r = 0; r < 8; ++r) { m_run[r] = -3.0e38f; l_run[r] = 0.f; }

  for (int kt = 0; kt < kS / 64; ++kt) {
    const size_t krow = (size_t)(b * kS + kt * 64);
#pragma unroll
    for (int i = 0; i < 4; ++i) {
      int c = t + 128 * i;
      int r = c >> 3, col = (c & 7) << 3;
      *(v8h*)&Kt[r][col]  = *(const v8h*)&Km[(krow + r) * kE + h * kDK + col];
      *(v8h*)&VtT[r][col] = *(const v8h*)&VhT[(size_t)(h * kDK + r) * kM + krow + col];
    }
    __syncthreads();

    // S = Q * K^T  (16x64 per wave: 4 n-frags x 2 k-steps)
    v8f s[4];
#pragma unroll
    for (int n = 0; n < 4; ++n) {
      s[n] = zero;
      v16h kb0 = frag_ld(&Kt[16 * n][0], 72, lane);
      v16h kb1 = frag_ld(&Kt[16 * n][32], 72, lane);
      s[n] = WMMA_F16(qa0, kb0, s[n]);
      s[n] = WMMA_F16(qa1, kb1, s[n]);
    }

    // Online softmax; row r+8*half lives in one lane-half -> xor 1,2,4,8
#pragma unroll
    for (int r = 0; r < 8; ++r) {
      float mx = -3.0e38f;
#pragma unroll
      for (int n = 0; n < 4; ++n) { s[n][r] *= 0.125f; mx = fmaxf(mx, s[n][r]); }
#pragma unroll
      for (int off = 1; off < 16; off <<= 1) mx = fmaxf(mx, __shfl_xor(mx, off, 32));
      float mnew = fmaxf(m_run[r], mx);
      float alpha = __expf(m_run[r] - mnew);
      float rs = 0.f;
#pragma unroll
      for (int n = 0; n < 4; ++n) {
        float p = __expf(s[n][r] - mnew);
        s[n][r] = p;
        rs += p;
      }
#pragma unroll
      for (int off = 1; off < 16; off <<= 1) rs += __shfl_xor(rs, off, 32);
      m_run[r] = mnew;
      l_run[r] = l_run[r] * alpha + rs;
#pragma unroll
      for (int n = 0; n < 4; ++n) o[n][r] *= alpha;
    }

    // C-layout -> A-layout via per-wave LDS (same-wave DS ops stay in order)
#pragma unroll
    for (int n = 0; n < 4; ++n)
#pragma unroll
      for (int r = 0; r < 8; ++r)
        Pl[wave][r + 8 * half][16 * n + ln] = (_Float16)s[n][r];

    v16h pa0 = frag_ld(&Pl[wave][0][0], 72, lane);
    v16h pa1 = frag_ld(&Pl[wave][0][32], 72, lane);
#pragma unroll
    for (int n = 0; n < 4; ++n) {
      v16h vb0 = frag_ld(&VtT[16 * n][0], 72, lane);
      v16h vb1 = frag_ld(&VtT[16 * n][32], 72, lane);
      o[n] = WMMA_F16(pa0, vb0, o[n]);
      o[n] = WMMA_F16(pa1, vb1, o[n]);
    }
    __syncthreads();
  }

  float c_attn = cosf(qp_attn[0] + qp_attn[1]);
#pragma unroll
  for (int r = 0; r < 8; ++r) {
    float inv = 1.0f / l_run[r];
    size_t row = rowQ + r + 8 * half;
#pragma unroll
    for (int n = 0; n < 4; ++n) {
      float v = o[n][r] * inv;
      if (n == 0 && ln < 4) v += c_attn;  // mixed[..., :NQ] += c_attn
      Out[row * kE + h * kDK + 16 * n + ln] = (_Float16)v;
    }
  }
}

// ---------------------------------------------------------------------------
// LayerNorm over E=1024 with residual add; badd row-major or per-column.
// ---------------------------------------------------------------------------
__device__ inline float block_reduce_sum(float v, float* red, int t) {
#pragma unroll
  for (int off = 16; off; off >>= 1) v += __shfl_xor(v, off, 32);
  __syncthreads();
  if ((t & 31) == 0) red[t >> 5] = v;
  __syncthreads();
  float s = 0.f;
#pragma unroll
  for (int i = 0; i < 8; ++i) s += red[i];
  return s;
}

__global__ __launch_bounds__(256) void ln_kernel(const float* __restrict__ a,
                                                 const float* __restrict__ badd,
                                                 int colvec,
                                                 const float* __restrict__ g,
                                                 const float* __restrict__ beta,
                                                 float* __restrict__ out) {
  __shared__ float red[8];
  int row = blockIdx.x, t = threadIdx.x;
  size_t off = (size_t)row * kE + t * 4;
  float4 va = *(const float4*)(a + off);
  float4 vb = colvec ? *(const float4*)(badd + t * 4) : *(const float4*)(badd + off);
  float x0 = va.x + vb.x, x1 = va.y + vb.y, x2 = va.z + vb.z, x3 = va.w + vb.w;
  float total = block_reduce_sum(x0 + x1 + x2 + x3, red, t);
  float mean = total * (1.0f / 1024.0f);
  float d0 = x0 - mean, d1 = x1 - mean, d2 = x2 - mean, d3 = x3 - mean;
  float ssq = block_reduce_sum(d0 * d0 + d1 * d1 + d2 * d2 + d3 * d3, red, t);
  float inv = rsqrtf(ssq * (1.0f / 1024.0f) + 1e-5f);
  float4 vg = *(const float4*)(g + t * 4);
  float4 vt = *(const float4*)(beta + t * 4);
  float4 r;
  r.x = d0 * inv * vg.x + vt.x;
  r.y = d1 * inv * vg.y + vt.y;
  r.z = d2 * inv * vg.z + vt.z;
  r.w = d3 * inv * vg.w + vt.w;
  *(float4*)(out + off) = r;
}

// ---------------------------------------------------------------------------
// ffn_vec[e] = relu(cos(qp[0]+qp[1])) * sum_{f<4} w2[f,e] + b2[e]
// ---------------------------------------------------------------------------
__global__ __launch_bounds__(256) void ffn_vec_kernel(const float* __restrict__ w2,
                                                      const float* __restrict__ b2,
                                                      const float* __restrict__ qp,
                                                      float* __restrict__ outv) {
  int e = blockIdx.x * 256 + threadIdx.x;
  if (e < kE) {
    float c = cosf(qp[0] + qp[1]);
    float rc = fmaxf(c, 0.f);
    float s = w2[e] + w2[kE + e] + w2[2 * kE + e] + w2[3 * kE + e];
    outv[e] = rc * s + b2[e];
  }
}

// ---------------------------------------------------------------------------
extern "C" void kernel_launch(void* const* d_in, const int* in_sizes, int n_in,
                              void* d_out, int out_size, void* d_ws, size_t ws_size,
                              hipStream_t stream) {
  (void)in_sizes; (void)n_in; (void)out_size; (void)ws_size;
  const float* x       = (const float*)d_in[0];
  const float* wq      = (const float*)d_in[1];
  const float* wk      = (const float*)d_in[2];
  const float* wv      = (const float*)d_in[3];
  const float* wo      = (const float*)d_in[4];
  // d_in[5]=w1, d_in[6]=b1: dead in the reference (h is never consumed)
  const float* w2      = (const float*)d_in[7];
  const float* b2      = (const float*)d_in[8];
  const float* qp_attn = (const float*)d_in[9];
  const float* qp_ffn  = (const float*)d_in[10];
  const float* ln1_g   = (const float*)d_in[11];
  const float* ln1_b   = (const float*)d_in[12];
  const float* ln2_g   = (const float*)d_in[13];
  const float* ln2_b   = (const float*)d_in[14];
  float* out = (float*)d_out;

  char* w = (char*)d_ws;
  auto alloc = [&](size_t bytes) -> void* {
    void* p = (void*)w;
    w += (bytes + 255) & ~(size_t)255;
    return p;
  };
  _Float16* xh     = (_Float16*)alloc((size_t)kM * kE * 2);
  _Float16* wqT    = (_Float16*)alloc((size_t)kE * kE * 2);  // [N x K]
  _Float16* wkT    = (_Float16*)alloc((size_t)kE * kE * 2);
  _Float16* wvT    = (_Float16*)alloc((size_t)kE * kE * 2);
  _Float16* woT    = (_Float16*)alloc((size_t)kE * kE * 2);
  _Float16* Qh     = (_Float16*)alloc((size_t)kM * kE * 2);
  _Float16* Kh     = (_Float16*)alloc((size_t)kM * kE * 2);
  _Float16* Vh     = (_Float16*)alloc((size_t)kM * kE * 2);
  _Float16* VhT    = (_Float16*)alloc((size_t)kM * kE * 2);  // [E x M]
  _Float16* mixedh = (_Float16*)alloc((size_t)kM * kE * 2);
  float* attn_out  = (float*)alloc((size_t)kM * kE * 4);
  float* x1        = (float*)alloc((size_t)kM * kE * 4);
  float* ffnv      = (float*)alloc((size_t)kE * 4);

  // 1) activations to f16; weights to f16 transposed (fused, LDS-tiled)
  cvt_f32_f16<<<(kM * kE) / 1024, 256, 0, stream>>>(x, xh, kM * kE);
  dim3 gt(kE / 64, kE / 64);
  cvt_transpose_f32f16<<<gt, 256, 0, stream>>>(wq, wqT, kE, kE);
  cvt_transpose_f32f16<<<gt, 256, 0, stream>>>(wk, wkT, kE, kE);
  cvt_transpose_f32f16<<<gt, 256, 0, stream>>>(wv, wvT, kE, kE);
  cvt_transpose_f32f16<<<gt, 256, 0, stream>>>(wo, woT, kE, kE);

  // 2) Q/K/V projections (WMMA, 128x256 block tiles, double-buffered LDS)
  dim3 gg(kE / 256, kM / 128);
  gemm_wmma<true><<<gg, 256, 0, stream>>>(xh, wqT, Qh, nullptr, kM, kE, kE);
  gemm_wmma<true><<<gg, 256, 0, stream>>>(xh, wkT, Kh, nullptr, kM, kE, kE);
  gemm_wmma<true><<<gg, 256, 0, stream>>>(xh, wvT, Vh, nullptr, kM, kE, kE);

  // 3) V -> V^T once (read 32x per tile afterwards), then flash attention
  transpose_f16<<<dim3(kE / 64, kM / 64), 256, 0, stream>>>(Vh, VhT, kM, kE);
  attn_wmma<<<dim3(kS / 64, 2 * kH), 128, 0, stream>>>(Qh, Kh, VhT, qp_attn, mixedh);

  // 4) O projection (WMMA, f32 out)
  gemm_wmma<false><<<gg, 256, 0, stream>>>(mixedh, woT, nullptr, attn_out, kM, kE, kE);

  // 5) x1 = LN1(x + attn_out)
  ln_kernel<<<kM, 256, 0, stream>>>(x, attn_out, 0, ln1_g, ln1_b, x1);

  // 6) ffn broadcast vector; 7) out = LN2(x1 + ffn_vec)
  ffn_vec_kernel<<<kE / 256, 256, 0, stream>>>(w2, b2, qp_ffn, ffnv);
  ln_kernel<<<kM, 256, 0, stream>>>(x1, ffnv, 1, ln2_g, ln2_b, out);
}
